// Attention_81458349736162
// MI455X (gfx1250) — matrix operand order repaired and checked by reading, not compiled, not run on hardware
//
#include <hip/hip_runtime.h>
#include <hip/hip_bf16.h>

// Problem constants (match reference)
#define B_   32
#define N_   4096
#define H_   512
#define OUT_ 5
#define MID_ 128

// ---------- types for WMMA ----------
typedef __bf16          v16bf  __attribute__((ext_vector_type(16)));
typedef unsigned short  v16u   __attribute__((ext_vector_type(16)));
typedef float           v8f    __attribute__((ext_vector_type(8)));
typedef __attribute__((address_space(3))) unsigned short* lds_us_ptr;

union BFragU { v16u u; v16bf b; };

// round-to-nearest-even f32 -> bf16 bits
__device__ __forceinline__ unsigned short f2bf(float x) {
  unsigned int u = __float_as_uint(x);
  u += 0x7FFFu + ((u >> 16) & 1u);
  return (unsigned short)(u >> 16);
}

// ----------------------------------------------------------------------------
// Kernel 0: pre-swizzle Wa_w (H x H, row-major (o,h)) into bf16 B-fragments.
// Layout (16-bit B 32x16, contiguous k order, not the ISA table order): lane l -> column n = l&15,
// k = 32*ks + (l>>4)*16 + e, e = 0..15 contiguous per lane.
// frag[((ot*16 + ks)*32 + lane)*16 + e]
// ----------------------------------------------------------------------------
__global__ __launch_bounds__(256) void k_convert(const float* __restrict__ Wa,
                                                 unsigned short* __restrict__ frag) {
  const int gid  = blockIdx.x * 256 + threadIdx.x;   // [0, 32*16*32)
  const int ot   = gid >> 9;          // 32 o-tiles
  const int rem  = gid & 511;
  const int ks   = rem >> 5;          // 16 k-steps
  const int lane = rem & 31;
  const int o    = ot * 16 + (lane & 15);
  const int kb   = ks * 32 + (lane >> 4) * 16;
  unsigned short* dst = frag + ((size_t)(ot * 16 + ks) * 32 + lane) * 16;
  const float*    src = Wa + (size_t)o * H_ + kb;
#pragma unroll
  for (int e = 0; e < 16; ++e) dst[e] = f2bf(src[e]);
}

// ----------------------------------------------------------------------------
// Kernel 1: cvec[b,o] = (Ua_w @ normal[b] + Ua_b)[o] - Wa_b[o]
// ----------------------------------------------------------------------------
__global__ __launch_bounds__(512) void k_ua(const float* __restrict__ nrm,
                                            const float* __restrict__ Uw,
                                            const float* __restrict__ Ub,
                                            const float* __restrict__ Wb,
                                            float* __restrict__ cvec) {
  __shared__ float nl[H_];
  const int b = blockIdx.x, t = threadIdx.x;
  nl[t] = nrm[b * H_ + t];
  __syncthreads();
  float a = Ub[t];
  const float* ur = Uw + (size_t)t * H_;
  for (int h = 0; h < H_; ++h) a = fmaf(ur[h], nl[h], a);
  cvec[b * H_ + t] = a - Wb[t];
}

// ----------------------------------------------------------------------------
// Kernel 2: distances via bf16 WMMA.
// Block = 128 threads = 4 waves; wave owns 16 rows of n (block tile = 64 rows).
// A tile (16 x 512) register-resident in bf16 (16 v16bf frags).
// B o-tiles (16 KB each) staged in LDS by global_load_async_to_lds_b128,
// double-buffered one tile ahead (ASYNCcnt), consumed via ds_load_b128.
// Two WMMA accumulators break the D->C dependency chain.
// ----------------------------------------------------------------------------
__global__ __launch_bounds__(128, 1) void k_dist(const float* __restrict__ defect,
                                                 const unsigned short* __restrict__ wfrag,
                                                 const float* __restrict__ cvec,
                                                 float* __restrict__ dist) {
  __shared__ unsigned short bstage[2][8192];   // 2 x 16KB B-tile buffers

  const int tid   = threadIdx.x;
  const int lane  = tid & 31;
  const int wave  = tid >> 5;
  const int b     = blockIdx.y;
  const int mbase = blockIdx.x * 64 + wave * 16;
  const int kh    = lane >> 4;        // which 16-lane half
  const int mrow  = lane & 15;        // A-matrix row owned by this lane

  // ---- kick async DMA of B tile 0 into buffer 0 (overlaps A load/convert)
  // tile = 1024 x 16B chunks; 128 threads x 8 chunks; per-lane LDS dest addr.
#define ISSUE_TILE(TILE, BUF)                                                   \
  {                                                                             \
    _Pragma("unroll")                                                           \
    for (int j = 0; j < 8; ++j) {                                               \
      const int chunk = j * 128 + tid;                                          \
      unsigned ldsa = (unsigned)(size_t)(lds_us_ptr)&bstage[BUF][chunk * 8];    \
      unsigned goff = (unsigned)((TILE) * 16384u + (unsigned)chunk * 16u);      \
      asm volatile("global_load_async_to_lds_b128 %0, %1, %2"                   \
                   :: "v"(ldsa), "v"(goff), "s"(wfrag) : "memory");             \
    }                                                                           \
  }

  ISSUE_TILE(0, 0)

  // ---- load + convert A tile (16-bit A 16x32, contiguous k to match B frags):
  // lane l, M = l&15; elements 0..15 -> K = 32*ks + kh*16 + e
  const float* rowp = defect + ((size_t)b * N_ + (mbase + mrow)) * H_;
  v16bf afrag[16];
#pragma unroll
  for (int ks = 0; ks < 16; ++ks) {
    const int k0 = ks * 32 + kh * 16;
    float4 p0 = *(const float4*)(rowp + k0);
    float4 p1 = *(const float4*)(rowp + k0 + 4);
    float4 q0 = *(const float4*)(rowp + k0 + 8);
    float4 q1 = *(const float4*)(rowp + k0 + 12);
    BFragU t;
    t.u[0]  = f2bf(p0.x); t.u[1]  = f2bf(p0.y); t.u[2]  = f2bf(p0.z); t.u[3]  = f2bf(p0.w);
    t.u[4]  = f2bf(p1.x); t.u[5]  = f2bf(p1.y); t.u[6]  = f2bf(p1.z); t.u[7]  = f2bf(p1.w);
    t.u[8]  = f2bf(q0.x); t.u[9]  = f2bf(q0.y); t.u[10] = f2bf(q0.z); t.u[11] = f2bf(q0.w);
    t.u[12] = f2bf(q1.x); t.u[13] = f2bf(q1.y); t.u[14] = f2bf(q1.z); t.u[15] = f2bf(q1.w);
    afrag[ks] = t.b;
  }

  float d2[8];
#pragma unroll
  for (int r = 0; r < 8; ++r) d2[r] = 0.0f;

  const float* cb = cvec + b * H_;

  for (int ot = 0; ot < 32; ++ot) {
    const int cur = ot & 1;
    // my async loads for tile `ot` complete (in-order), then all waves' ones
    asm volatile("s_wait_asynccnt 0" ::: "memory");
    __syncthreads();
    // prefetch next tile into the other buffer; that buffer was last read in
    // iteration ot-1, and every wave is past that compute (barrier above).
    if (ot + 1 < 32) ISSUE_TILE(ot + 1, cur ^ 1)

    // consume tile from LDS: frag(ks,lane) at ushort offset (ks*32+lane)*16
    const v16u* sp = (const v16u*)&bstage[cur][lane * 16];
    v8f acc0 = {}, acc1 = {};
#pragma unroll
    for (int ks = 0; ks < 16; ks += 2) {
      BFragU bf0, bf1;
      bf0.u = sp[(ks    ) * 32];
      bf1.u = sp[(ks + 1) * 32];
      acc0 = __builtin_amdgcn_wmma_f32_16x16x32_bf16(
          false, afrag[ks],     false, bf0.b, (short)0, acc0, false, false);
      acc1 = __builtin_amdgcn_wmma_f32_16x16x32_bf16(
          false, afrag[ks + 1], false, bf1.b, (short)0, acc1, false, false);
    }
    const v8f acc = acc0 + acc1;

    // epilogue: C/D layout -> VGPR r: rows r (lanes 0-15) / r+8 (lanes 16-31),
    // column o = ot*16 + (lane&15) in both halves.
    const float c = cb[ot * 16 + mrow];
#pragma unroll
    for (int r = 0; r < 8; ++r) { float d = acc[r] - c; d2[r] = fmaf(d, d, d2[r]); }
  }
#undef ISSUE_TILE

  // reduce over the 16 columns within each lane-half
#pragma unroll
  for (int r = 0; r < 8; ++r) {
    float s = d2[r];
    s += __shfl_xor(s, 1);
    s += __shfl_xor(s, 2);
    s += __shfl_xor(s, 4);
    s += __shfl_xor(s, 8);
    if (mrow == 0)  // lane 0 -> row r, lane 16 -> row r+8
      dist[(size_t)b * N_ + mbase + r + kh * 8] = sqrtf(s);
  }
}

// ----------------------------------------------------------------------------
// Kernel 3: softmax over N per batch
// ----------------------------------------------------------------------------
__global__ __launch_bounds__(256) void k_softmax(const float* __restrict__ dist,
                                                 float* __restrict__ w) {
  __shared__ float red[256];
  const int b = blockIdx.x, t = threadIdx.x;
  const float* d = dist + (size_t)b * N_;
  float mx = -1e30f;
  for (int n = t; n < N_; n += 256) mx = fmaxf(mx, d[n]);
  red[t] = mx; __syncthreads();
  for (int s = 128; s > 0; s >>= 1) { if (t < s) red[t] = fmaxf(red[t], red[t + s]); __syncthreads(); }
  mx = red[0]; __syncthreads();
  float sum = 0.0f;
  for (int n = t; n < N_; n += 256) sum += __expf(d[n] - mx);
  red[t] = sum; __syncthreads();
  for (int s = 128; s > 0; s >>= 1) { if (t < s) red[t] += red[t + s]; __syncthreads(); }
  const float inv = 1.0f / red[0];
  for (int n = t; n < N_; n += 256) w[(size_t)b * N_ + n] = __expf(d[n] - mx) * inv;
}

// ----------------------------------------------------------------------------
// Kernel 4: context[b,h] = sum_n w[b,n] * defect[b,n,h]  (coalesced over h)
// ----------------------------------------------------------------------------
__global__ __launch_bounds__(512) void k_context(const float* __restrict__ w,
                                                 const float* __restrict__ defect,
                                                 float* __restrict__ ctx) {
  __shared__ float wl[512];
  const int b = blockIdx.x, t = threadIdx.x;
  float acc = 0.0f;
  for (int n0 = 0; n0 < N_; n0 += 512) {
    __syncthreads();
    wl[t] = w[(size_t)b * N_ + n0 + t];
    __syncthreads();
    const float* dp = defect + ((size_t)b * N_ + n0) * H_ + t;
    if (n0 + 512 < N_)  // warm GL2 for the next chunk (global_prefetch_b8)
      __builtin_prefetch(defect + ((size_t)b * N_ + n0 + 512) * H_ + t, 0, 0);
    for (int j = 0; j < 512; ++j) acc = fmaf(wl[j], dp[(size_t)j * H_], acc);
  }
  ctx[b * H_ + t] = acc;
}

// ----------------------------------------------------------------------------
// Kernel 5: MLP head  relu([ctx, gf] @ W1^T + b1) @ W2^T + b2
// ----------------------------------------------------------------------------
__global__ __launch_bounds__(128) void k_mlp(const float* __restrict__ ctx,
                                             const float* __restrict__ gf,
                                             const float* __restrict__ W1,
                                             const float* __restrict__ b1,
                                             const float* __restrict__ W2,
                                             const float* __restrict__ b2,
                                             float* __restrict__ out) {
  __shared__ float comb[2 * H_];
  __shared__ float h1[MID_];
  const int b = blockIdx.x, t = threadIdx.x;
  for (int i = t; i < H_; i += 128) {
    comb[i]       = ctx[b * H_ + i];
    comb[H_ + i]  = gf[b * H_ + i];
  }
  __syncthreads();
  float a = b1[t];
  const float* wr = W1 + (size_t)t * (2 * H_);
  for (int j = 0; j < 2 * H_; ++j) a = fmaf(wr[j], comb[j], a);
  h1[t] = fmaxf(a, 0.0f);
  __syncthreads();
  if (t < OUT_) {
    float o = b2[t];
    const float* w2r = W2 + (size_t)t * MID_;
    for (int j = 0; j < MID_; ++j) o = fmaf(w2r[j], h1[j], o);
    out[b * OUT_ + t] = o;
  }
}

// ----------------------------------------------------------------------------
extern "C" void kernel_launch(void* const* d_in, const int* in_sizes, int n_in,
                              void* d_out, int out_size, void* d_ws, size_t ws_size,
                              hipStream_t stream) {
  (void)in_sizes; (void)n_in; (void)out_size; (void)ws_size;

  const float* nrm    = (const float*)d_in[0];   // [B,1,H]
  const float* defect = (const float*)d_in[1];   // [B,N,H]
  const float* gf     = (const float*)d_in[2];   // [B,H]
  const float* Wa_w   = (const float*)d_in[3];   // [H,H]
  const float* Wa_b   = (const float*)d_in[4];   // [H]
  const float* Ua_w   = (const float*)d_in[5];   // [H,H]
  const float* Ua_b   = (const float*)d_in[6];   // [H]
  const float* W1     = (const float*)d_in[7];   // [MID,2H]
  const float* b1     = (const float*)d_in[8];   // [MID]
  const float* W2     = (const float*)d_in[9];   // [OUT,MID]
  const float* b2     = (const float*)d_in[10];  // [OUT]
  float* out = (float*)d_out;

  char* ws = (char*)d_ws;
  unsigned short* frag = (unsigned short*)(ws);            // 512 KB bf16 B-fragments
  float* cvec = (float*)(ws + 524288);                     //  64 KB
  float* dist = (float*)(ws + 589824);                     // 512 KB
  float* wts  = (float*)(ws + 1114112);                    // 512 KB
  float* ctx  = (float*)(ws + 1638400);                    //  64 KB  (total ~1.63 MB)

  k_convert<<<64, 256, 0, stream>>>(Wa_w, frag);
  k_ua<<<B_, 512, 0, stream>>>(nrm, Ua_w, Ua_b, Wa_b, cvec);
  k_dist<<<dim3(N_ / 64, B_), 128, 0, stream>>>(defect, frag, cvec, dist);
  k_softmax<<<B_, 256, 0, stream>>>(dist, wts);
  k_context<<<B_, 512, 0, stream>>>(wts, defect, ctx);
  k_mlp<<<B_, 128, 0, stream>>>(ctx, gf, W1, b1, W2, b2, out);
}
